// MambaFFN_88167088652785
// MI455X (gfx1250) — matrix operand order repaired and checked by reading, not compiled, not run on hardware
//
#include <hip/hip_runtime.h>
#include <math.h>

// Sizes (compile-time constants from the reference)
#define Bb     4
#define Ll     2048
#define Dd     64
#define D_IN   128
#define D_ST   64
#define DT_R   4
#define FF_IN  256
#define NTOK   (Bb * Ll)       // 8192
#define XDB_C  (DT_R + 2*D_ST) // 132
#define PFF    (Ll + 3)        // 2051 conv output length per batch
#define NC     32              // scan chunks
#define LC     (Ll / NC)       // 64 steps per chunk

typedef __attribute__((ext_vector_type(16))) _Float16 v16h;
typedef __attribute__((ext_vector_type(8)))  float    v8f;

#if defined(__has_builtin)
#if __has_builtin(__builtin_amdgcn_s_wait_tensorcnt)
#define WAIT_TENSOR() __builtin_amdgcn_s_wait_tensorcnt(0)
#endif
#endif
#ifndef WAIT_TENSOR
#define WAIT_TENSOR()
#endif

// WMMA 16x16x32 f16 fragment K-offset patterns (CDNA5 ISA 7.12.2)
__device__ __forceinline__ int a_koff(int i, int half) {
    return ((i < 4) ? (2 * i) : (16 + 2 * (i - 4))) + 8 * half;
}
__device__ __forceinline__ int b_koff(int i, int half) {
    return ((i < 4) ? (2 * i) : (16 + 2 * (i - 4))) + 8 * half;
}

// ---------------------------------------------------------------------------
// K1: xz(8192,256) = x(8192,64) @ in_proj_w(256,64)^T        [WMMA, K=64]
// ---------------------------------------------------------------------------
__global__ void k_in_proj(const float* __restrict__ x,
                          const float* __restrict__ w,
                          float* __restrict__ xz) {
    int wave = (blockIdx.x * blockDim.x + threadIdx.x) >> 5;
    int lane = threadIdx.x & 31;
    const int NT = 2 * D_IN / 16;           // 16 column tiles
    int mt = wave / NT, nt = wave % NT;
    int mrow = lane & 15, half = lane >> 4;
    int m0 = mt * 16, n0 = nt * 16;
    v8f acc = {};
    for (int kc = 0; kc < 2; ++kc) {
        int k0 = kc * 32;
        v16h a, b;
        const float* arow = x + (m0 + mrow) * Dd + k0;
        const float* bcol = w + (n0 + mrow) * Dd + k0;
#pragma unroll
        for (int i = 0; i < 8; ++i) {
            int ka = a_koff(i, half);
            a[2*i]   = (_Float16)arow[ka];
            a[2*i+1] = (_Float16)arow[ka + 1];
            int kb = b_koff(i, half);
            b[2*i]   = (_Float16)bcol[kb];
            b[2*i+1] = (_Float16)bcol[kb + 1];
        }
        acc = __builtin_amdgcn_wmma_f32_16x16x32_f16(false, a, false, b,
                                                     (short)0, acc, false, false);
    }
#pragma unroll
    for (int r = 0; r < 8; ++r)
        xz[(m0 + r + 8*half) * (2*D_IN) + n0 + mrow] = acc[r];
}

// ---------------------------------------------------------------------------
// K2: causal depthwise conv (K=4) over xin = xz[:, :128], + bias, SiLU
// ---------------------------------------------------------------------------
__global__ void k_conv(const float* __restrict__ xz,
                       const float* __restrict__ cw,
                       const float* __restrict__ cb,
                       float* __restrict__ xc) {
    int idx = blockIdx.x * blockDim.x + threadIdx.x;   // 8192*128
    int c = idx & (D_IN - 1);
    int m = idx >> 7;
    int l = m & (Ll - 1);
    float acc = cb[c];
#pragma unroll
    for (int j = 0; j < 4; ++j) {
        int lp = l - 3 + j;
        if (lp >= 0) acc += xz[(m - l + lp) * (2*D_IN) + c] * cw[c * 4 + j];
    }
    xc[idx] = acc / (1.f + __expf(-acc));              // SiLU
}

// ---------------------------------------------------------------------------
// K3: xdb(8192,132) = xc(8192,128) @ x_proj_w(132,128)^T     [WMMA, K=128]
// ---------------------------------------------------------------------------
__global__ void k_x_proj(const float* __restrict__ xc,
                         const float* __restrict__ w,
                         float* __restrict__ xdb) {
    int wave = (blockIdx.x * blockDim.x + threadIdx.x) >> 5;
    int lane = threadIdx.x & 31;
    int nt = wave % 9, mt = wave / 9;                  // 9 N-tiles (132 cols)
    int mrow = lane & 15, half = lane >> 4;
    int m0 = mt * 16, n0 = nt * 16;
    int ncol = n0 + mrow;
    bool nok = ncol < XDB_C;
    v8f acc = {};
    for (int kc = 0; kc < 4; ++kc) {
        int k0 = kc * 32;
        v16h a, b;
        const float* arow = xc + (m0 + mrow) * D_IN + k0;
        const float* bcol = w + ncol * D_IN + k0;
#pragma unroll
        for (int i = 0; i < 8; ++i) {
            int ka = a_koff(i, half);
            a[2*i]   = (_Float16)arow[ka];
            a[2*i+1] = (_Float16)arow[ka + 1];
            int kb = b_koff(i, half);
            b[2*i]   = (_Float16)(nok ? bcol[kb]     : 0.f);
            b[2*i+1] = (_Float16)(nok ? bcol[kb + 1] : 0.f);
        }
        acc = __builtin_amdgcn_wmma_f32_16x16x32_f16(false, a, false, b,
                                                     (short)0, acc, false, false);
    }
#pragma unroll
    for (int r = 0; r < 8; ++r)
        if (nok) xdb[(m0 + r + 8*half) * XDB_C + ncol] = acc[r];
}

// ---------------------------------------------------------------------------
// K4: dt(8192,128) = softplus(dtr @ dt_proj_w^T + b)         [tiny K=4, VALU]
// ---------------------------------------------------------------------------
__global__ void k_dt(const float* __restrict__ xdb,
                     const float* __restrict__ w,
                     const float* __restrict__ bias,
                     float* __restrict__ dtg) {
    int idx = blockIdx.x * blockDim.x + threadIdx.x;   // 8192*128
    int d = idx & (D_IN - 1);
    int m = idx >> 7;
    float acc = bias[d];
#pragma unroll
    for (int r = 0; r < 4; ++r)
        acc += xdb[m * XDB_C + r] * w[d * 4 + r];
    dtg[idx] = acc > 20.f ? acc : log1pf(__expf(acc));
}

// ---------------------------------------------------------------------------
// Chunked parallel scan (3 phases).  h_t = dA_t*h_{t-1} + dBx_t is
// associative in (a,b), so: phase1 computes per-chunk (P=prod dA, E=local
// end state) with 512-way block parallelism; phase2 composes the 32 chunk
// transforms serially (tiny); phase3 replays each chunk from its true
// incoming state H_in and emits the gated output.
// ---------------------------------------------------------------------------
__global__ void k_scan_phase1(const float* __restrict__ xdb,
                              const float* __restrict__ xc,
                              const float* __restrict__ dtg,
                              const float* __restrict__ A_log,
                              float* __restrict__ Pbuf,
                              float* __restrict__ Ebuf) {
    __shared__ float sB[D_ST], sdt[32], sx[32];
    int dchunk = blockIdx.x & 3;
    int c = (blockIdx.x >> 2) & (NC - 1);
    int b = blockIdx.x >> 7;
    int tid = threadIdx.x;
    int dloc = tid >> 3;
    int d = dchunk * 32 + dloc;
    int sbase = (tid & 7) * 8;
    float A[8], h[8], pA[8];
#pragma unroll
    for (int j = 0; j < 8; ++j) {
        A[j] = -__expf(A_log[d * D_ST + sbase + j]);
        h[j] = 0.f;
        pA[j] = 1.f;
    }
    for (int t = c * LC; t < (c + 1) * LC; ++t) {
        int m = b * Ll + t;
        if (tid < 64)       sB[tid]       = xdb[m * XDB_C + DT_R + tid];
        else if (tid < 96)  sdt[tid - 64] = dtg[m * D_IN + dchunk * 32 + (tid - 64)];
        else if (tid < 128) sx[tid - 96]  = xc[m * D_IN + dchunk * 32 + (tid - 96)];
        __syncthreads();
        float dtv = sdt[dloc], xv = sx[dloc];
#pragma unroll
        for (int j = 0; j < 8; ++j) {
            float dA = __expf(dtv * A[j]);
            h[j] = dA * h[j] + dtv * sB[sbase + j] * xv;
            pA[j] *= dA;
        }
        __syncthreads();
    }
    int base = ((b * NC + c) * D_IN + d) * D_ST + sbase;
#pragma unroll
    for (int j = 0; j < 8; ++j) {
        Pbuf[base + j] = pA[j];
        Ebuf[base + j] = h[j];
    }
}

__global__ void k_scan_combine(float* __restrict__ Pbuf,
                               float* __restrict__ Ebuf) {
    int dchunk = blockIdx.x & 3;
    int b = blockIdx.x >> 2;
    int tid = threadIdx.x;
    int dloc = tid >> 3;
    int d = dchunk * 32 + dloc;
    int sbase = (tid & 7) * 8;
    float H[8];
#pragma unroll
    for (int j = 0; j < 8; ++j) H[j] = 0.f;
    for (int c = 0; c < NC; ++c) {
        int base = ((b * NC + c) * D_IN + d) * D_ST + sbase;
#pragma unroll
        for (int j = 0; j < 8; ++j) {
            float P = Pbuf[base + j];
            float E = Ebuf[base + j];
            Ebuf[base + j] = H[j];          // H_in for this chunk
            H[j] = P * H[j] + E;
        }
    }
}

__global__ void k_scan_phase3(const float* __restrict__ xz,
                              const float* __restrict__ xc,
                              const float* __restrict__ xdb,
                              const float* __restrict__ dtg,
                              const float* __restrict__ A_log,
                              const float* __restrict__ Dp,
                              const float* __restrict__ Hin,
                              float* __restrict__ yg) {
    __shared__ float sB[D_ST], sC[D_ST], sdt[32], sx[32];
    int dchunk = blockIdx.x & 3;
    int c = (blockIdx.x >> 2) & (NC - 1);
    int b = blockIdx.x >> 7;
    int tid = threadIdx.x;
    int dloc = tid >> 3;
    int d = dchunk * 32 + dloc;
    int sbase = (tid & 7) * 8;
    int base = ((b * NC + c) * D_IN + d) * D_ST + sbase;
    float A[8], h[8];
#pragma unroll
    for (int j = 0; j < 8; ++j) {
        A[j] = -__expf(A_log[d * D_ST + sbase + j]);
        h[j] = Hin[base + j];
    }
    float Dv = Dp[d];
    for (int t = c * LC; t < (c + 1) * LC; ++t) {
        int m = b * Ll + t;
        if (tid < 64)        sB[tid]        = xdb[m * XDB_C + DT_R + tid];
        else if (tid < 128)  sC[tid - 64]   = xdb[m * XDB_C + DT_R + D_ST + (tid - 64)];
        else if (tid < 160)  sdt[tid - 128] = dtg[m * D_IN + dchunk * 32 + (tid - 128)];
        else if (tid < 192)  sx[tid - 160]  = xc[m * D_IN + dchunk * 32 + (tid - 160)];
        __syncthreads();
        float dtv = sdt[dloc], xv = sx[dloc];
        float p = 0.f;
#pragma unroll
        for (int j = 0; j < 8; ++j) {
            float dA = __expf(dtv * A[j]);
            h[j] = dA * h[j] + dtv * sB[sbase + j] * xv;
            p += h[j] * sC[sbase + j];
        }
        p += __shfl_xor(p, 1, 32);
        p += __shfl_xor(p, 2, 32);
        p += __shfl_xor(p, 4, 32);
        if ((tid & 7) == 0) {
            float z = xz[m * (2*D_IN) + D_IN + d];
            yg[m * D_IN + d] = (p + xv * Dv) * (z / (1.f + __expf(-z)));
        }
        __syncthreads();
    }
}

// ---------------------------------------------------------------------------
// K6: h(8192,64) = yg @ out_proj_w^T  [WMMA K=128] + fused leakyReLU +
//     group RMSNorm (16-ch group == one N tile; 4x shfl_xor within halves)
// ---------------------------------------------------------------------------
__global__ void k_out_proj_norm(const float* __restrict__ yg,
                                const float* __restrict__ w,
                                const float* __restrict__ gamma,
                                float* __restrict__ hnorm) {
    int wave = (blockIdx.x * blockDim.x + threadIdx.x) >> 5;
    int lane = threadIdx.x & 31;
    int mt = wave >> 2, nt = wave & 3;      // 4 N-tiles (64 cols)
    int mrow = lane & 15, half = lane >> 4;
    int m0 = mt * 16, n0 = nt * 16;
    v8f acc = {};
    for (int kc = 0; kc < 4; ++kc) {
        int k0 = kc * 32;
        v16h a, b;
        const float* arow = yg + (m0 + mrow) * D_IN + k0;
        const float* bcol = w + (n0 + mrow) * D_IN + k0;
#pragma unroll
        for (int i = 0; i < 8; ++i) {
            int ka = a_koff(i, half);
            a[2*i]   = (_Float16)arow[ka];
            a[2*i+1] = (_Float16)arow[ka + 1];
            int kb = b_koff(i, half);
            b[2*i]   = (_Float16)bcol[kb];
            b[2*i+1] = (_Float16)bcol[kb + 1];
        }
        acc = __builtin_amdgcn_wmma_f32_16x16x32_f16(false, a, false, b,
                                                     (short)0, acc, false, false);
    }
    float g = gamma[n0 + mrow];
#pragma unroll
    for (int r = 0; r < 8; ++r) {
        float v = acc[r];
        v = v > 0.f ? v : 0.01f * v;        // leaky_relu
        float ss = v * v;                   // row-wise sum over the 16 group
        ss += __shfl_xor(ss, 1, 32);        // lanes (16-lane halves hold one
        ss += __shfl_xor(ss, 2, 32);        // matrix row each)
        ss += __shfl_xor(ss, 4, 32);
        ss += __shfl_xor(ss, 8, 32);
        float rms = sqrtf(ss) * 0.25f;      // * DPG^-0.5
        hnorm[(m0 + r + 8*half) * Dd + n0 + mrow] = v / (rms + 1e-5f) * g;
    }
}

// ---------------------------------------------------------------------------
// K7: FF conv as im2col GEMM, K=256 (c,k). One 512-thread block per
//     (batch, M-tile): the 19x64 activation slab is staged once to LDS (f16)
//     and shared by all 16 N-tile waves; each wave computes the paired u/g
//     tiles reusing its A fragments (16 WMMAs/wave), fused bias + SwiGLU.
// ---------------------------------------------------------------------------
__global__ void k_ff_conv(const float* __restrict__ hnorm,
                          const float* __restrict__ w,
                          const float* __restrict__ bias,
                          float* __restrict__ hmid) {
    __shared__ _Float16 sH[19 * 72];        // 19 rows x 64 ch, padded stride
    int mt = blockIdx.x % 129;              // position tile (2051 -> 129)
    int b  = blockIdx.x / 129;
    int p0 = mt * 16;
    const float* hb = hnorm + b * Ll * Dd;
    for (int idx = threadIdx.x; idx < 19 * Dd; idx += blockDim.x) {
        int r = idx >> 6, c = idx & 63;
        int lp = p0 - 3 + r;
        float v = (lp >= 0 && lp < Ll) ? hb[lp * Dd + c] : 0.f;
        sH[r * 72 + c] = (_Float16)v;
    }
    __syncthreads();
    int nt   = threadIdx.x >> 5;            // 16 waves = 16 u-channel tiles
    int lane = threadIdx.x & 31;
    int mrow = lane & 15, half = lane >> 4;
    int ou = nt * 16 + mrow;                // B-col channel for this lane
    __builtin_prefetch(w + ou * (Dd * 4), 0, 3);
    __builtin_prefetch(w + (FF_IN + ou) * (Dd * 4), 0, 3);
    v8f accU = {}, accG = {};
    for (int kc = 0; kc < 8; ++kc) {        // K = 256 = 8 chunks
        int kk0 = kc * 32;
        v16h a, bu, bg;
#pragma unroll
        for (int i = 0; i < 8; ++i) {
            int kk = kk0 + a_koff(i, half);
            int k = kk >> 6, c = kk & 63;   // kk = k*64 + c
            a[2*i]   = sH[(mrow + k) * 72 + c];      // row p0+mrow+k-3
            a[2*i+1] = sH[(mrow + k) * 72 + c + 1];
            int kb = kk0 + b_koff(i, half);
            int k2 = kb >> 6, c2 = kb & 63;
            bu[2*i]   = (_Float16)w[ou * (Dd*4) + c2 * 4 + k2];
            bu[2*i+1] = (_Float16)w[ou * (Dd*4) + (c2 + 1) * 4 + k2];
            bg[2*i]   = (_Float16)w[(FF_IN + ou) * (Dd*4) + c2 * 4 + k2];
            bg[2*i+1] = (_Float16)w[(FF_IN + ou) * (Dd*4) + (c2 + 1) * 4 + k2];
        }
        accU = __builtin_amdgcn_wmma_f32_16x16x32_f16(false, a, false, bu,
                                                      (short)0, accU, false, false);
        accG = __builtin_amdgcn_wmma_f32_16x16x32_f16(false, a, false, bg,
                                                      (short)0, accG, false, false);
    }
    float bU = bias[ou];
    float bG = bias[FF_IN + ou];
#pragma unroll
    for (int r = 0; r < 8; ++r) {
        int pr = p0 + r + 8 * half;
        if (pr < PFF) {
            float u = accU[r] + bU;
            float gg = accG[r] + bG;
            float sil = gg / (1.f + __expf(-gg));
            hmid[(b * PFF + pr) * FF_IN + ou] = u * sil;
        }
    }
}

// ---------------------------------------------------------------------------
// K8: FF deconv as GEMM, K=1024, A slab staged to LDS as f16 and shared by
//     4 waves (32 WMMAs/wave). Fused: out = hnorm + 0.5*(y + db)
// ---------------------------------------------------------------------------
__global__ void k_ff_deconv(const float* __restrict__ hmid,
                            const float* __restrict__ dw,
                            const float* __restrict__ db,
                            const float* __restrict__ hnorm,
                            float* __restrict__ out) {
    __shared__ _Float16 sA[19 * 264];       // 19 rows x 256 ch, padded stride
    WAIT_TENSOR();
    int b  = blockIdx.x >> 7;               // 128 M-tiles per batch
    int mt = blockIdx.x & 127;
    int t0 = mt * 16;
    const float* hm = hmid + b * PFF * FF_IN;
    __builtin_prefetch(dw, 0, 3);
    for (int idx = threadIdx.x; idx < 19 * FF_IN; idx += blockDim.x) {
        int r = idx >> 8, c = idx & 255;
        sA[r * 264 + c] = (_Float16)hm[(t0 + r) * FF_IN + c];
    }
    __syncthreads();
    int wave = threadIdx.x >> 5;            // = N tile (0..3)
    int lane = threadIdx.x & 31;
    int mrow = lane & 15, half = lane >> 4;
    int n0 = wave * 16;
    int d = n0 + mrow;
    v8f acc = {};
    for (int kc = 0; kc < 32; ++kc) {       // K = 1024, kk = k*256 + c
        int kk0 = kc * 32;
        v16h a, bb;
#pragma unroll
        for (int i = 0; i < 8; ++i) {
            int kk = kk0 + a_koff(i, half);
            int k = kk >> 8, c = kk & 255;
            a[2*i]   = sA[(mrow + k) * 264 + c];
            a[2*i+1] = sA[(mrow + k) * 264 + c + 1];
            int kb = kk0 + b_koff(i, half);
            int k2 = kb >> 8, c2 = kb & 255;
            bb[2*i]   = (_Float16)dw[c2 * (Dd*4) + d * 4 + (3 - k2)];
            bb[2*i+1] = (_Float16)dw[(c2 + 1) * (Dd*4) + d * 4 + (3 - k2)];
        }
        acc = __builtin_amdgcn_wmma_f32_16x16x32_f16(false, a, false, bb,
                                                     (short)0, acc, false, false);
    }
    float bd = db[d];
#pragma unroll
    for (int r = 0; r < 8; ++r) {
        int t = t0 + r + 8 * half;
        int gi = (b * Ll + t) * Dd + d;
        out[gi] = hnorm[gi] + 0.5f * (acc[r] + bd);
    }
}

// ---------------------------------------------------------------------------
extern "C" void kernel_launch(void* const* d_in, const int* in_sizes, int n_in,
                              void* d_out, int out_size, void* d_ws, size_t ws_size,
                              hipStream_t stream) {
    const float* x       = (const float*)d_in[0];
    const float* in_w    = (const float*)d_in[1];
    const float* conv_w  = (const float*)d_in[2];
    const float* conv_b  = (const float*)d_in[3];
    const float* xproj_w = (const float*)d_in[4];
    const float* dt_w    = (const float*)d_in[5];
    const float* dt_b    = (const float*)d_in[6];
    const float* A_log   = (const float*)d_in[7];
    const float* Dp      = (const float*)d_in[8];
    const float* out_w   = (const float*)d_in[9];
    const float* gamma   = (const float*)d_in[10];
    const float* ffw     = (const float*)d_in[11];
    const float* ffb     = (const float*)d_in[12];
    const float* dfw     = (const float*)d_in[13];
    const float* dfb     = (const float*)d_in[14];
    float* out = (float*)d_out;

    float* ws    = (float*)d_ws;
    float* xz    = ws;                              // 8192*256
    float* xc    = xz    + NTOK * 2 * D_IN;         // 8192*128
    float* xdb   = xc    + NTOK * D_IN;             // 8192*132
    float* dtbuf = xdb   + NTOK * XDB_C;            // 8192*128
    float* yg    = dtbuf + NTOK * D_IN;             // 8192*128
    float* hnorm = yg    + NTOK * D_IN;             // 8192*64
    float* hmid  = hnorm + NTOK * Dd;               // 4*2051*256
    // Scan scratch (P, E/Hin) aliases hmid: consumed by phase3 before K7
    // writes hmid.  Each is B*NC*D_IN*D_ST = 1,048,576 floats; hmid holds
    // 2,100,224 floats, enough for both.
    float* Pbuf  = hmid;
    float* Ebuf  = hmid + Bb * NC * D_IN * D_ST;

    // K1: in_proj GEMM (8192/16 * 256/16 = 8192 waves -> 1024 blocks)
    k_in_proj<<<1024, 256, 0, stream>>>(x, in_w, xz);
    // K2: causal conv + SiLU (8192*128 threads)
    k_conv<<<4096, 256, 0, stream>>>(xz, conv_w, conv_b, xc);
    // K3: x_proj GEMM (512 * 9 = 4608 waves -> 576 blocks)
    k_x_proj<<<576, 256, 0, stream>>>(xc, xproj_w, xdb);
    // K4: dt softplus
    k_dt<<<4096, 256, 0, stream>>>(xdb, dt_w, dt_b, dtbuf);
    // K5: chunked scan: 4*32*4 = 512 blocks, then 16, then 512
    k_scan_phase1<<<512, 256, 0, stream>>>(xdb, xc, dtbuf, A_log, Pbuf, Ebuf);
    k_scan_combine<<<16, 256, 0, stream>>>(Pbuf, Ebuf);
    k_scan_phase3<<<512, 256, 0, stream>>>(xz, xc, xdb, dtbuf, A_log, Dp, Ebuf, yg);
    // K6: out_proj + leaky + groupRMSNorm (512*4 = 2048 waves -> 256 blocks)
    k_out_proj_norm<<<256, 256, 0, stream>>>(yg, out_w, gamma, hnorm);
    // K7: FF conv + SwiGLU (4*129 blocks x 512 threads)
    k_ff_conv<<<516, 512, 0, stream>>>(hnorm, ffw, ffb, hmid);
    // K8: FF deconv + residual (4*128 blocks, 4 waves each)
    k_ff_deconv<<<512, 128, 0, stream>>>(hmid, dfw, dfb, hnorm, out);
}
